// get_model_19567871000891
// MI455X (gfx1250) — matrix operand order repaired and checked by reading, not compiled, not run on hardware
//
#include <hip/hip_runtime.h>

// ---------------------------------------------------------------------------
// PointNet++ forward for MI455X (gfx1250, wave32).
// GEMM layers run on v_wmma_f32_16x16x32_bf16 with 32x32 per-wave tiles
// (4 WMMA per K-step, 2 b128 loads per WMMA); irregular ops are scalar kernels.
// ---------------------------------------------------------------------------

#define Bn 16
#define Nn 4096

typedef __bf16 bf16;
typedef bf16  v16bf __attribute__((ext_vector_type(16)));
typedef float v8f   __attribute__((ext_vector_type(8)));

union Frag { v16bf v; unsigned int d[8]; };

static __device__ __forceinline__ unsigned short f2bf(float f) {
  unsigned int u = __float_as_uint(f);
  unsigned int r = u + 0x7FFFu + ((u >> 16) & 1u);   // round-to-nearest-even
  return (unsigned short)(r >> 16);
}

// --------------------------- data movement kernels -------------------------

// xyz [B,9,N] -> coords [B*N,3], feats [B*N,6]
__global__ void k_split(const float* __restrict__ xyz, float* __restrict__ coords,
                        float* __restrict__ feats) {
  int idx = blockIdx.x * blockDim.x + threadIdx.x;
  if (idx >= Bn * Nn) return;
  int b = idx / Nn, n = idx % Nn;
  const float* src = xyz + (size_t)b * 9 * Nn + n;
#pragma unroll
  for (int i = 0; i < 3; ++i) coords[(size_t)idx * 3 + i] = src[(size_t)i * Nn];
#pragma unroll
  for (int i = 0; i < 6; ++i) feats[(size_t)idx * 6 + i] = src[(size_t)(3 + i) * Nn];
}

// f32 [R,C] -> bf16 [R,Kp] zero-padded
__global__ void k_cvt_pad(const float* __restrict__ src, unsigned short* __restrict__ dst,
                          int R, int C, int Kp) {
  size_t total = (size_t)R * Kp;
  for (size_t i = blockIdx.x * (size_t)blockDim.x + threadIdx.x; i < total;
       i += (size_t)gridDim.x * blockDim.x) {
    int c = (int)(i % Kp);
    size_t r = i / Kp;
    dst[i] = (c < C) ? f2bf(src[r * C + c]) : 0;
  }
}

// W f32 [Co,Ci] -> bf16 [Cop,Kp] zero-padded both dims
__global__ void k_cvt_w(const float* __restrict__ W, unsigned short* __restrict__ dst,
                        int Co, int Ci, int Cop, int Kp) {
  size_t total = (size_t)Cop * Kp;
  for (size_t i = blockIdx.x * (size_t)blockDim.x + threadIdx.x; i < total;
       i += (size_t)gridDim.x * blockDim.x) {
    int c = (int)(i % Kp);
    int r = (int)(i / Kp);
    dst[i] = (r < Co && c < Ci) ? f2bf(W[(size_t)r * Ci + c]) : 0;
  }
}

// concat(a[R,Ca], b[R,Cb]) -> bf16 [R,Kp] zero-padded
__global__ void k_concat_cvt(const float* __restrict__ a, int Ca,
                             const float* __restrict__ bsrc, int Cb,
                             unsigned short* __restrict__ dst, int R, int Kp) {
  size_t total = (size_t)R * Kp;
  for (size_t i = blockIdx.x * (size_t)blockDim.x + threadIdx.x; i < total;
       i += (size_t)gridDim.x * blockDim.x) {
    int c = (int)(i % Kp);
    size_t r = i / Kp;
    float v = 0.f;
    if (c < Ca)            v = a[r * Ca + c];
    else if (c < Ca + Cb)  v = bsrc[r * Cb + (c - Ca)];
    dst[i] = f2bf(v);
  }
}

// ------------------------------- WMMA GEMM ---------------------------------
// Y[R,Co] = Xb[R,Kp](bf16) * Wb[Cop,Kp]^T(bf16) + bias.
// One wave = 32x32 output tile (2 A-frags x 2 B-frags -> 4 WMMA / K-step).
// Block = 4 waves in 2x2 -> 64x64 block tile. Cop is always a multiple of 32.

static __device__ __forceinline__ void load_a(Frag& f, const unsigned short* row,
                                              int k0, int half) {
#pragma unroll
  for (int v = 0; v < 8; ++v) {
    // A 16x32 bf16 layout: vgpr v holds K = {0|16} + half*8 + (v&3)*2 (+1)
    int ka = k0 + ((v < 4) ? 0 : 16) + half * 8 + (v & 3) * 2;
    f.d[v] = *(const unsigned int*)(row + ka);
  }
}
static __device__ __forceinline__ void load_b(Frag& f, const unsigned short* row,
                                              int k0, int half) {
#pragma unroll
  for (int v = 0; v < 8; ++v) {
    // B 32x16 bf16 layout: vgpr v holds K = {0|16} + half*8 + (v&3)*2 (+1), N = lane&15
    int kb = k0 + ((v < 4) ? 0 : 16) + half * 8 + (v & 3) * 2;
    f.d[v] = *(const unsigned int*)(row + kb);
  }
}
static __device__ __forceinline__ void store_c(const v8f& acc, float* __restrict__ Y,
                                               int m0, int n0, int R, int Co,
                                               int half, int l15,
                                               const float* __restrict__ bias) {
  int cn = n0 + l15;
  if (cn >= Co) return;
  float bv = bias[cn];
#pragma unroll
  for (int i = 0; i < 8; ++i) {       // C/D: vgpr i -> M = i + 8*half
    int cm = m0 + i + 8 * half;
    if (cm < R) Y[(size_t)cm * Co + cn] = acc[i] + bv;
  }
}

__global__ void k_gemm(const unsigned short* __restrict__ Xb,
                       const unsigned short* __restrict__ Wb,
                       const float* __restrict__ bias, float* __restrict__ Y,
                       int R, int Kp, int Co, int Cop) {
  int lane = threadIdx.x & 31;
  int wave = threadIdx.x >> 5;       // 0..3
  int half = lane >> 4;
  int l15 = lane & 15;
  int m0 = blockIdx.x * 64 + (wave & 1) * 32;
  int n0 = blockIdx.y * 64 + (wave >> 1) * 32;
  if (n0 >= Cop) return;

  int ra0 = m0 + l15;      if (ra0 >= R) ra0 = R - 1;   // clamp; stores guarded
  int ra1 = m0 + 16 + l15; if (ra1 >= R) ra1 = R - 1;
  const unsigned short* A0 = Xb + (size_t)ra0 * Kp;
  const unsigned short* A1 = Xb + (size_t)ra1 * Kp;
  const unsigned short* B0 = Wb + (size_t)(n0 + l15) * Kp;        // Cop mult of 32
  const unsigned short* B1 = Wb + (size_t)(n0 + 16 + l15) * Kp;

  v8f z = {0.f, 0.f, 0.f, 0.f, 0.f, 0.f, 0.f, 0.f};
  v8f acc00 = z, acc01 = z, acc10 = z, acc11 = z;
  for (int k0 = 0; k0 < Kp; k0 += 32) {
    Frag a0, a1, b0, b1;
    load_a(a0, A0, k0, half);
    load_a(a1, A1, k0, half);
    load_b(b0, B0, k0, half);
    load_b(b1, B1, k0, half);
    acc00 = __builtin_amdgcn_wmma_f32_16x16x32_bf16(false, a0.v, false, b0.v, (short)0, acc00, false, false);
    acc01 = __builtin_amdgcn_wmma_f32_16x16x32_bf16(false, a0.v, false, b1.v, (short)0, acc01, false, false);
    acc10 = __builtin_amdgcn_wmma_f32_16x16x32_bf16(false, a1.v, false, b0.v, (short)0, acc10, false, false);
    acc11 = __builtin_amdgcn_wmma_f32_16x16x32_bf16(false, a1.v, false, b1.v, (short)0, acc11, false, false);
  }
  store_c(acc00, Y, m0,      n0,      R, Co, half, l15, bias);
  store_c(acc01, Y, m0,      n0 + 16, R, Co, half, l15, bias);
  store_c(acc10, Y, m0 + 16, n0,      R, Co, half, l15, bias);
  store_c(acc11, Y, m0 + 16, n0 + 16, R, Co, half, l15, bias);
}

// ---------------------------- BatchNorm pieces -----------------------------

__global__ void k_bn_stats(const float* __restrict__ Y, float* __restrict__ mean,
                           float* __restrict__ rstd, int R, int C) {
  int c = blockIdx.x;
  float s = 0.f, ss = 0.f;
  for (int r = threadIdx.x; r < R; r += blockDim.x) {
    float v = Y[(size_t)r * C + c];
    s += v; ss += v * v;
  }
  __shared__ float sh[256], sh2[256];
  sh[threadIdx.x] = s; sh2[threadIdx.x] = ss;
  __syncthreads();
  for (int o = 128; o > 0; o >>= 1) {
    if ((int)threadIdx.x < o) { sh[threadIdx.x] += sh[threadIdx.x + o];
                                sh2[threadIdx.x] += sh2[threadIdx.x + o]; }
    __syncthreads();
  }
  if (threadIdx.x == 0) {
    float m = sh[0] / R;
    float var = sh2[0] / R - m * m;
    mean[c] = m;
    rstd[c] = rsqrtf(var + 1e-5f);
  }
}

// BN+ReLU; optional f32 out [R,C]; optional bf16 out [R,Kp] (padded)
__global__ void k_bn_relu(const float* __restrict__ Y, const float* __restrict__ mean,
                          const float* __restrict__ rstd, const float* __restrict__ g,
                          const float* __restrict__ beta, float* __restrict__ outF,
                          unsigned short* __restrict__ outB, int R, int C, int Kp) {
  size_t total = (size_t)R * Kp;
  for (size_t i = blockIdx.x * (size_t)blockDim.x + threadIdx.x; i < total;
       i += (size_t)gridDim.x * blockDim.x) {
    int c = (int)(i % Kp);
    size_t r = i / Kp;
    float v = 0.f;
    if (c < C) {
      v = (Y[r * C + c] - mean[c]) * rstd[c] * g[c] + beta[c];
      v = fmaxf(v, 0.f);
      if (outF) outF[r * C + c] = v;
    }
    if (outB) outB[i] = f2bf(v);
  }
}

// STN: BN+ReLU then max over N: Y [B,N,C] -> out [B,C]
__global__ void k_bn_relu_maxpool(const float* __restrict__ Y, const float* __restrict__ mean,
                                  const float* __restrict__ rstd, const float* __restrict__ g,
                                  const float* __restrict__ beta, float* __restrict__ out, int C) {
  int idx = blockIdx.x * blockDim.x + threadIdx.x;
  if (idx >= Bn * C) return;
  int b = idx / C, c = idx % C;
  float m = mean[c], a = rstd[c] * g[c], be = beta[c];
  const float* p = Y + (size_t)b * Nn * C + c;
  float mx = 0.f;   // ReLU output >= 0
  for (int n = 0; n < Nn; ++n) {
    float v = (p[(size_t)n * C] - m) * a + be;
    v = fmaxf(v, 0.f);
    mx = fmaxf(mx, v);
  }
  out[idx] = mx;
}

// SA: BN+ReLU then max over nsample groups: Y [BS*ns, C] -> out [BS, C]
__global__ void k_bn_relu_maxg(const float* __restrict__ Y, const float* __restrict__ mean,
                               const float* __restrict__ rstd, const float* __restrict__ g,
                               const float* __restrict__ beta, float* __restrict__ out,
                               int BS, int ns, int C) {
  int idx = blockIdx.x * blockDim.x + threadIdx.x;
  if (idx >= BS * C) return;
  int bs = idx / C, c = idx % C;
  float m = mean[c], a = rstd[c] * g[c], be = beta[c];
  const float* p = Y + ((size_t)bs * ns) * C + c;
  float mx = 0.f;
  for (int j = 0; j < ns; ++j) {
    float v = (p[(size_t)j * C] - m) * a + be;
    v = fmaxf(v, 0.f);
    mx = fmaxf(mx, v);
  }
  out[idx] = mx;
}

// --------------------------- point-set operations --------------------------

// farthest point sampling; one block per batch, sequential over S samples
__global__ void k_fps(const float* __restrict__ xyz, int* __restrict__ fidx,
                      float* __restrict__ dist, int Nin, int S) {
  int b = blockIdx.x;
  const float* P = xyz + (size_t)b * Nin * 3;
  float* D = dist + (size_t)b * Nin;
  for (int n = threadIdx.x; n < Nin; n += blockDim.x) D[n] = 1e10f;
  __shared__ float sv[256];
  __shared__ int   si[256];
  __shared__ int   sfar;
  if (threadIdx.x == 0) sfar = 0;
  __syncthreads();
  for (int it = 0; it < S; ++it) {
    int far = sfar;
    if (threadIdx.x == 0) fidx[b * S + it] = far;
    float cx = P[far * 3], cy = P[far * 3 + 1], cz = P[far * 3 + 2];
    float bv = -1.f; int bi = 0;
    for (int n = threadIdx.x; n < Nin; n += blockDim.x) {
      float dx = P[n * 3] - cx, dy = P[n * 3 + 1] - cy, dz = P[n * 3 + 2] - cz;
      float d = dx * dx + dy * dy + dz * dz;
      float dd = fminf(D[n], d);
      D[n] = dd;
      if (dd > bv) { bv = dd; bi = n; }
    }
    sv[threadIdx.x] = bv; si[threadIdx.x] = bi;
    __syncthreads();
    for (int o = 128; o > 0; o >>= 1) {
      if ((int)threadIdx.x < o && sv[threadIdx.x + o] > sv[threadIdx.x]) {
        sv[threadIdx.x] = sv[threadIdx.x + o]; si[threadIdx.x] = si[threadIdx.x + o];
      }
      __syncthreads();
    }
    if (threadIdx.x == 0) sfar = si[0];
    __syncthreads();
  }
}

__global__ void k_gather(const float* __restrict__ xyz, const int* __restrict__ fidx,
                         float* __restrict__ out, int Nin, int S) {
  int idx = blockIdx.x * blockDim.x + threadIdx.x;
  if (idx >= Bn * S) return;
  int b = idx / S;
  int n = fidx[idx];
  const float* p = xyz + ((size_t)b * Nin + n) * 3;
  out[(size_t)idx * 3 + 0] = p[0];
  out[(size_t)idx * 3 + 1] = p[1];
  out[(size_t)idx * 3 + 2] = p[2];
}

// ball query: ascending-index selection == reference's sort-then-truncate
__global__ void k_ballq(const float* __restrict__ xyz, const float* __restrict__ cxyz,
                        int* __restrict__ gidx, int Nin, int S, float r2) {
  int idx = blockIdx.x * blockDim.x + threadIdx.x;
  if (idx >= Bn * S) return;
  int b = idx / S;
  const float* P = xyz + (size_t)b * Nin * 3;
  float cx = cxyz[(size_t)idx * 3], cy = cxyz[(size_t)idx * 3 + 1], cz = cxyz[(size_t)idx * 3 + 2];
  int* out = gidx + (size_t)idx * 32;
  int cnt = 0, first = 0;
  for (int n = 0; n < Nin && cnt < 32; ++n) {
    float dx = P[n * 3] - cx, dy = P[n * 3 + 1] - cy, dz = P[n * 3 + 2] - cz;
    if (dx * dx + dy * dy + dz * dz <= r2) {
      if (cnt == 0) first = n;
      out[cnt++] = n;
    }
  }
  for (; cnt < 32; ++cnt) out[cnt] = first;
}

// build grouped bf16 GEMM input: [(xyz-center) | feats] zero-padded to Kp
__global__ void k_group(const float* __restrict__ xyz, const float* __restrict__ cxyz,
                        const float* __restrict__ feats, const int* __restrict__ gidx,
                        unsigned short* __restrict__ out, int Nin, int S, int D, int Kp) {
  int idx = blockIdx.x * blockDim.x + threadIdx.x;
  if (idx >= Bn * S * 32) return;
  int bs = idx / 32;
  int b = bs / S;
  int n = gidx[idx];
  const float* p = xyz + ((size_t)b * Nin + n) * 3;
  const float* c = cxyz + (size_t)bs * 3;
  unsigned short* o = out + (size_t)idx * Kp;
  o[0] = f2bf(p[0] - c[0]);
  o[1] = f2bf(p[1] - c[1]);
  o[2] = f2bf(p[2] - c[2]);
  const float* f = feats + ((size_t)b * Nin + n) * D;
  for (int d = 0; d < D; ++d) o[3 + d] = f2bf(f[d]);
  for (int d = 3 + D; d < Kp; ++d) o[d] = 0;
}

// 3-NN inverse-distance interpolation
__global__ void k_interp(const float* __restrict__ x1, const float* __restrict__ x2,
                         const float* __restrict__ p2, float* __restrict__ out,
                         int N1, int N2, int C) {
  int idx = blockIdx.x * blockDim.x + threadIdx.x;
  if (idx >= Bn * N1) return;
  int b = idx / N1;
  float ax = x1[(size_t)idx * 3], ay = x1[(size_t)idx * 3 + 1], az = x1[(size_t)idx * 3 + 2];
  const float* Q = x2 + (size_t)b * N2 * 3;
  float d0 = 1e30f, d1 = 1e30f, d2 = 1e30f;
  int i0 = 0, i1 = 0, i2 = 0;
  for (int m = 0; m < N2; ++m) {
    float dx = Q[m * 3] - ax, dy = Q[m * 3 + 1] - ay, dz = Q[m * 3 + 2] - az;
    float d = dx * dx + dy * dy + dz * dz;
    if (d < d0)      { d2 = d1; i2 = i1; d1 = d0; i1 = i0; d0 = d; i0 = m; }
    else if (d < d1) { d2 = d1; i2 = i1; d1 = d; i1 = m; }
    else if (d < d2) { d2 = d; i2 = m; }
  }
  float w0 = 1.f / (d0 + 1e-8f), w1 = 1.f / (d1 + 1e-8f), w2 = 1.f / (d2 + 1e-8f);
  float ws = w0 + w1 + w2;
  w0 /= ws; w1 /= ws; w2 /= ws;
  const float* F = p2 + (size_t)b * N2 * C;
  float* O = out + (size_t)idx * C;
  for (int c = 0; c < C; ++c)
    O[c] = w0 * F[(size_t)i0 * C + c] + w1 * F[(size_t)i1 * C + c] + w2 * F[(size_t)i2 * C + c];
}

// ----------------------------- small tails ---------------------------------

__global__ void k_linear(const float* __restrict__ X, const float* __restrict__ W,
                         const float* __restrict__ bias, float* __restrict__ Y,
                         int R, int Ci, int Co) {
  int idx = blockIdx.x * blockDim.x + threadIdx.x;
  if (idx >= R * Co) return;
  int r = idx / Co, o = idx % Co;
  float acc = bias[o];
  const float* x = X + (size_t)r * Ci;
  const float* w = W + (size_t)o * Ci;
  for (int c = 0; c < Ci; ++c) acc += x[c] * w[c];
  Y[idx] = acc;
}

// fp1b: [B,N,64] -> logits written transposed into d_out [B,13,N]
__global__ void k_fp1b(const float* __restrict__ X, const float* __restrict__ W,
                       const float* __restrict__ bias, float* __restrict__ out) {
  int idx = blockIdx.x * blockDim.x + threadIdx.x;
  if (idx >= Bn * Nn * 13) return;
  int o = idx % 13;
  int rn = idx / 13;
  int b = rn / Nn, n = rn % Nn;
  float acc = bias[o];
  const float* x = X + (size_t)rn * 64;
  const float* w = W + (size_t)o * 64;
#pragma unroll
  for (int c = 0; c < 64; ++c) acc += x[c] * w[c];
  out[((size_t)b * 13 + o) * Nn + n] = acc;
}

__global__ void k_make_trans(const float* __restrict__ h, float* __restrict__ tws,
                             float* __restrict__ tout) {
  int idx = blockIdx.x * blockDim.x + threadIdx.x;
  if (idx >= Bn * 9) return;
  int k = idx % 9;
  float v = h[idx] + ((k % 4 == 0) ? 1.f : 0.f);   // +I on diagonal (k=0,4,8)
  tws[idx] = v;
  tout[idx] = v;
}

__global__ void k_apply_trans(const float* __restrict__ T, const float* __restrict__ coords,
                              float* __restrict__ out) {
  int idx = blockIdx.x * blockDim.x + threadIdx.x;
  if (idx >= Bn * Nn) return;
  int b = idx / Nn;
  const float* t = T + (size_t)b * 9;
  float x = coords[(size_t)idx * 3], y = coords[(size_t)idx * 3 + 1], z = coords[(size_t)idx * 3 + 2];
#pragma unroll
  for (int i = 0; i < 3; ++i)
    out[(size_t)idx * 3 + i] = t[i * 3] * x + t[i * 3 + 1] * y + t[i * 3 + 2] * z;
}

// ------------------------------- host side ---------------------------------

struct Lin { const float *W, *b, *beta, *g; int Ci, Co; };

static inline int pad16(int x) { return (x + 15) & ~15; }
static inline int pad32(int x) { return (x + 31) & ~31; }
static inline unsigned eb(size_t n) { return (unsigned)((n + 255) / 256); }
static inline unsigned gs(size_t n) { size_t b = (n + 255) / 256; return (unsigned)(b > 65535 ? 65535 : b); }

extern "C" void kernel_launch(void* const* d_in, const int* in_sizes, int n_in,
                              void* d_out, int out_size, void* d_ws, size_t ws_size,
                              hipStream_t stream) {
  (void)in_sizes; (void)n_in; (void)out_size; (void)ws_size;
  auto F = [&](int i) { return (const float*)d_in[i]; };

  // jax pytree order (sorted dict keys): params{fp1a,fp1b,fp2,fp3,fp4,sa1..4,stn}, xyz
  // each BN-lin dict flattens as W, b, beta, g
  Lin fp1a{F(0),  F(1),  F(2),  F(3),  70,  64};
  const float *fp1bW = F(4), *fp1bB = F(5);
  Lin fp2 {F(6),  F(7),  F(8),  F(9),  192, 64};
  Lin fp3 {F(10), F(11), F(12), F(13), 384, 128};
  Lin fp4 {F(14), F(15), F(16), F(17), 768, 256};
  Lin sa1m[3] = { {F(18),F(19),F(20),F(21),9,32},   {F(22),F(23),F(24),F(25),32,32},   {F(26),F(27),F(28),F(29),32,64} };
  Lin sa2m[3] = { {F(30),F(31),F(32),F(33),67,64},  {F(34),F(35),F(36),F(37),64,64},   {F(38),F(39),F(40),F(41),64,128} };
  Lin sa3m[3] = { {F(42),F(43),F(44),F(45),131,128},{F(46),F(47),F(48),F(49),128,128}, {F(50),F(51),F(52),F(53),128,256} };
  Lin sa4m[3] = { {F(54),F(55),F(56),F(57),259,256},{F(58),F(59),F(60),F(61),256,256}, {F(62),F(63),F(64),F(65),256,512} };
  Lin c1{F(66),F(67),F(68),F(69),3,64}, c2{F(70),F(71),F(72),F(73),64,128}, c3{F(74),F(75),F(76),F(77),128,1024};
  Lin f1{F(78),F(79),F(80),F(81),1024,512}, f2{F(82),F(83),F(84),F(85),512,256};
  const float *f3W = F(86), *f3B = F(87);
  const float* xyz = F(88);

  // workspace bump allocator
  char* wp = (char*)d_ws;
  auto alloc = [&](size_t bytes) -> void* {
    void* p = wp; wp += (bytes + 255) & ~(size_t)255; return p;
  };
  const int R0 = Bn * Nn;
  float* coords = (float*)alloc(sizeof(float) * R0 * 3);
  float* feats  = (float*)alloc(sizeof(float) * R0 * 6);
  float* l0xyz  = (float*)alloc(sizeof(float) * R0 * 3);
  float* transw = (float*)alloc(sizeof(float) * Bn * 9);
  unsigned short* Xb = (unsigned short*)alloc(2ull * 16777216);     // 33.5 MB bf16 staging
  float* Y = (float*)alloc(4ull * 67108864);                        // 268 MB GEMM output
  unsigned short* wb = (unsigned short*)alloc(2ull * 524288);       // weight staging
  float* meanb = (float*)alloc(sizeof(float) * 1024);
  float* rstdb = (float*)alloc(sizeof(float) * 1024);
  float* distb = (float*)alloc(sizeof(float) * Bn * Nn);
  int* fidx = (int*)alloc(sizeof(int) * Bn * 1024);
  int* gidx = (int*)alloc(sizeof(int) * Bn * 1024 * 32);
  float* l1xyz = (float*)alloc(sizeof(float) * Bn * 1024 * 3);
  float* l2xyz = (float*)alloc(sizeof(float) * Bn * 256 * 3);
  float* l3xyz = (float*)alloc(sizeof(float) * Bn * 64 * 3);
  float* l4xyz = (float*)alloc(sizeof(float) * Bn * 16 * 3);
  float* l1f = (float*)alloc(sizeof(float) * Bn * 1024 * 64);
  float* l2f = (float*)alloc(sizeof(float) * Bn * 256 * 128);
  float* l3f = (float*)alloc(sizeof(float) * Bn * 64 * 256);
  float* l4f = (float*)alloc(sizeof(float) * Bn * 16 * 512);
  float* l3n = (float*)alloc(sizeof(float) * Bn * 64 * 256);
  float* l2n = (float*)alloc(sizeof(float) * Bn * 256 * 128);
  float* l1n = (float*)alloc(sizeof(float) * Bn * 1024 * 64);
  float* itp = (float*)alloc(sizeof(float) * Bn * Nn * 64);
  float* hpool = (float*)alloc(sizeof(float) * Bn * 1024);
  float* h2 = (float*)alloc(sizeof(float) * Bn * 256);
  float* hf3 = (float*)alloc(sizeof(float) * Bn * 9);
  float* l0f = (float*)alloc(sizeof(float) * (size_t)R0 * 64);

  // GEMM + BN-stats for one layer (Xb must hold [R, pad32(L.Ci)] bf16)
  auto gemm = [&](const Lin& L, int R) {
    int Kp = pad32(L.Ci), Cop = pad16(L.Co);
    k_cvt_w<<<gs((size_t)Cop * Kp), 256, 0, stream>>>(L.W, wb, L.Co, L.Ci, Cop, Kp);
    dim3 g((unsigned)((R + 63) / 64), (unsigned)((Cop + 63) / 64));
    k_gemm<<<g, 128, 0, stream>>>(Xb, wb, L.b, Y, R, Kp, L.Co, Cop);
    k_bn_stats<<<dim3((unsigned)L.Co), 256, 0, stream>>>(Y, meanb, rstdb, R, L.Co);
  };

  // ---------------- STN ----------------
  k_split<<<eb((size_t)R0), 256, 0, stream>>>(xyz, coords, feats);
  k_cvt_pad<<<gs((size_t)R0 * 32), 256, 0, stream>>>(coords, Xb, R0, 3, 32);
  gemm(c1, R0);
  k_bn_relu<<<gs((size_t)R0 * 64), 256, 0, stream>>>(Y, meanb, rstdb, c1.g, c1.beta, (float*)nullptr, Xb, R0, 64, 64);
  gemm(c2, R0);
  k_bn_relu<<<gs((size_t)R0 * 128), 256, 0, stream>>>(Y, meanb, rstdb, c2.g, c2.beta, (float*)nullptr, Xb, R0, 128, 128);
  gemm(c3, R0);
  k_bn_relu_maxpool<<<eb((size_t)Bn * 1024), 256, 0, stream>>>(Y, meanb, rstdb, c3.g, c3.beta, hpool, 1024);
  k_cvt_pad<<<gs((size_t)Bn * 1024), 256, 0, stream>>>(hpool, Xb, Bn, 1024, 1024);
  gemm(f1, Bn);
  k_bn_relu<<<gs((size_t)Bn * 512), 256, 0, stream>>>(Y, meanb, rstdb, f1.g, f1.beta, (float*)nullptr, Xb, Bn, 512, 512);
  gemm(f2, Bn);
  k_bn_relu<<<gs((size_t)Bn * 256), 256, 0, stream>>>(Y, meanb, rstdb, f2.g, f2.beta, h2, (unsigned short*)nullptr, Bn, 256, 256);
  k_linear<<<eb((size_t)Bn * 9), 256, 0, stream>>>(h2, f3W, f3B, hf3, Bn, 256, 9);
  k_make_trans<<<1, 256, 0, stream>>>(hf3, transw, (float*)d_out + (size_t)Bn * 13 * Nn);
  k_apply_trans<<<eb((size_t)R0), 256, 0, stream>>>(transw, coords, l0xyz);

  // ---------------- set abstraction ----------------
  auto sa = [&](const float* inx, int Nin, const float* inf, int D, int S, float radius,
                const Lin* mlp, float* outx, float* outf) {
    k_fps<<<Bn, 256, 0, stream>>>(inx, fidx, distb, Nin, S);
    k_gather<<<eb((size_t)Bn * S), 256, 0, stream>>>(inx, fidx, outx, Nin, S);
    k_ballq<<<eb((size_t)Bn * S), 256, 0, stream>>>(inx, outx, gidx, Nin, S, radius * radius);
    int R = Bn * S * 32;
    int Kp = pad32(3 + D);
    k_group<<<eb((size_t)R), 256, 0, stream>>>(inx, outx, inf, gidx, Xb, Nin, S, D, Kp);
    for (int i = 0; i < 3; ++i) {
      const Lin& L = mlp[i];
      gemm(L, R);
      if (i < 2) {
        int Kn = pad32(L.Co);
        k_bn_relu<<<gs((size_t)R * Kn), 256, 0, stream>>>(Y, meanb, rstdb, L.g, L.beta,
                                                          (float*)nullptr, Xb, R, L.Co, Kn);
      } else {
        k_bn_relu_maxg<<<eb((size_t)Bn * S * L.Co), 256, 0, stream>>>(Y, meanb, rstdb, L.g, L.beta,
                                                                      outf, Bn * S, 32, L.Co);
      }
    }
  };
  sa(l0xyz, Nn,   feats, 6,   1024, 0.1f, sa1m, l1xyz, l1f);
  sa(l1xyz, 1024, l1f,   64,  256,  0.2f, sa2m, l2xyz, l2f);
  sa(l2xyz, 256,  l2f,   128, 64,   0.4f, sa3m, l3xyz, l3f);
  sa(l3xyz, 64,   l3f,   256, 16,   0.8f, sa4m, l4xyz, l4f);

  // ---------------- feature propagation ----------------
  auto fp = [&](const float* x1, int N1, const float* x2, int N2, const float* p2, int C2,
                const float* skip, int C1, const Lin& L, float* outF) {
    k_interp<<<eb((size_t)Bn * N1), 256, 0, stream>>>(x1, x2, p2, itp, N1, N2, C2);
    int R = Bn * N1, Kp = pad32(C1 + C2);
    k_concat_cvt<<<gs((size_t)R * Kp), 256, 0, stream>>>(skip, C1, itp, C2, Xb, R, Kp);
    gemm(L, R);
    k_bn_relu<<<gs((size_t)R * L.Co), 256, 0, stream>>>(Y, meanb, rstdb, L.g, L.beta,
                                                        outF, (unsigned short*)nullptr, R, L.Co, L.Co);
  };
  fp(l3xyz, 64,   l4xyz, 16,   l4f, 512, l3f,   256, fp4,  l3n);
  fp(l2xyz, 256,  l3xyz, 64,   l3n, 256, l2f,   128, fp3,  l2n);
  fp(l1xyz, 1024, l2xyz, 256,  l2n, 128, l1f,   64,  fp2,  l1n);
  fp(l0xyz, Nn,   l1xyz, 1024, l1n, 64,  feats, 6,   fp1a, l0f);

  k_fp1b<<<eb((size_t)Bn * Nn * 13), 256, 0, stream>>>(l0f, fp1bW, fp1bB, (float*)d_out);
}
